// MambaLayer_19653770347250
// MI455X (gfx1250) — compile-verified, matrix operand order checked by reading, not run on hardware
//
#include <hip/hip_runtime.h>
#include <hip/hip_bf16.h>

// ---------------- problem constants ----------------
#define DMODEL 1024
#define DSTATE 16
#define DCONV  4
#define DINNER 2048
#define DTRANK 64
#define BB     2
#define LL     2048
#define MTOK   (BB * LL)             // 4096 tokens
#define DBLW   (DTRANK + 2 * DSTATE) // 96 (true width)
#define DBLP   128                   // padded to a multiple of 64 for the GEMM

// ---------------- types ----------------
typedef __attribute__((ext_vector_type(16))) __bf16 v16bf;
typedef __attribute__((ext_vector_type(8)))  float  v8f;

struct B32Pack { uint4 lo; uint4 hi; };  // 32 bytes -> one v16bf fragment

static __device__ __forceinline__ __bf16 f2bf(float f) {
  // round-to-nearest-even f32 -> bf16
  unsigned u = __builtin_bit_cast(unsigned, f);
  unsigned r = (u + 0x7FFFu + ((u >> 16) & 1u)) >> 16;
  unsigned short s = (unsigned short)r;
  return __builtin_bit_cast(__bf16, s);
}

static __device__ __forceinline__ v16bf load_frag32(const __bf16* p) {
  // ISA 16-bit operand layout: this lane's fragment is 8 bf16 at p[0..7]
  // (VGPR0-3) and 8 bf16 at p[16..23] (VGPR4-7).
  B32Pack pk;
  pk.lo = *reinterpret_cast<const uint4*>(p);
  pk.hi = *reinterpret_cast<const uint4*>(p + 16);
  return __builtin_bit_cast(v16bf, pk);
}

// ---------------- LayerNorm -> bf16 ----------------
__global__ void ln_kernel(const float* __restrict__ x, const float* __restrict__ w,
                          const float* __restrict__ b, __bf16* __restrict__ out) {
  const int token = blockIdx.x;               // 0..MTOK-1
  const float* xr = x + (size_t)token * DMODEL;
  float v[4];
  float s = 0.f, s2 = 0.f;
#pragma unroll
  for (int i = 0; i < 4; ++i) {
    v[i] = xr[threadIdx.x + i * 256];
    s += v[i]; s2 += v[i] * v[i];
  }
#pragma unroll
  for (int off = 16; off > 0; off >>= 1) {
    s  += __shfl_down(s, off);
    s2 += __shfl_down(s2, off);
  }
  __shared__ float ls[8], ls2[8];
  const int wid = threadIdx.x >> 5;
  if ((threadIdx.x & 31) == 0) { ls[wid] = s; ls2[wid] = s2; }
  __syncthreads();
  if (threadIdx.x == 0) {
    float S = 0.f, S2 = 0.f;
#pragma unroll
    for (int i = 0; i < 8; ++i) { S += ls[i]; S2 += ls2[i]; }
    ls[0] = S; ls2[0] = S2;
  }
  __syncthreads();
  const float mu  = ls[0] * (1.f / DMODEL);
  const float var = ls2[0] * (1.f / DMODEL) - mu * mu;
  const float inv = rsqrtf(var + 1e-5f);
#pragma unroll
  for (int i = 0; i < 4; ++i) {
    const int c = threadIdx.x + i * 256;
    out[(size_t)token * DMODEL + c] = f2bf((v[i] - mu) * inv * w[c] + b[c]);
  }
}

// ---------------- f32 -> bf16 convert (zero-pad beyond n_src) ----------------
__global__ void cvt_bf16_kernel(const float* __restrict__ src, __bf16* __restrict__ dst,
                                int n_src, int n_total) {
  const int i = blockIdx.x * blockDim.x + threadIdx.x;
  if (i >= n_total) return;
  const float f = (i < n_src) ? src[i] : 0.f;
  dst[i] = f2bf(f);
}

// ---------------- WMMA bf16 GEMM: C[M,N] = A[M,K] * W[N,K]^T (+ add) -------
// One wave computes a 16(M) x 64(N) tile: one A fragment reused across four
// B fragments per K-step of 32. Block = 256 threads = 8 waves.
// REQUIREMENTS: N % 64 == 0, M % 16 == 0, K % 32 == 0, grid sized exactly to
// (M/16)*(N/64)/8 blocks. No divergent branches anywhere -> EXEC all-1s for
// every WMMA, no accumulator copies/spills.
template <bool HAS_ADD>
__global__ void gemm_wmma_bf16(const __bf16* __restrict__ A, const __bf16* __restrict__ W,
                               float* __restrict__ C, const float* __restrict__ add,
                               int M, int N, int K) {
  const int lane = threadIdx.x & 31;
  // force wave index into an SGPR so tile math is provably wave-uniform
  const int wave = __builtin_amdgcn_readfirstlane(threadIdx.x >> 5);
  const int n64tiles = N >> 6;
  const int tile = blockIdx.x * 8 + wave;
  const int mt = tile / n64tiles;
  const int nt = tile - mt * n64tiles;
  const int m0 = mt << 4;
  const int n0 = nt << 6;

  // 16-bit operand layout: lane<16 holds K {0..7,16..23}; lane>=16 holds
  // K {8..15,24..31}; row/col = lane & 15.
  const int l15    = lane & 15;
  const int klocal = (lane >> 4) << 3;       // 0 or 8
  const __bf16* aRow  = A + (size_t)(m0 + l15) * K + klocal;
  const __bf16* bBase = W + (size_t)(n0 + l15) * K + klocal;

  v8f acc0 = {0.f}, acc1 = {0.f}, acc2 = {0.f}, acc3 = {0.f};
  const size_t bStride16 = (size_t)16 * K;   // 16 rows of W

  for (int k = 0; k < K; k += 32) {
    __builtin_prefetch(aRow + k + 32, 0, 1);           // global_prefetch path
    __builtin_prefetch(bBase + k + 32, 0, 1);
    const v16bf af = load_frag32(aRow + k);
    const v16bf b0 = load_frag32(bBase + k);
    const v16bf b1 = load_frag32(bBase + bStride16 + k);
    const v16bf b2 = load_frag32(bBase + 2 * bStride16 + k);
    const v16bf b3 = load_frag32(bBase + 3 * bStride16 + k);
    acc0 = __builtin_amdgcn_wmma_f32_16x16x32_bf16(false, af, false, b0, (short)0, acc0, false, false);
    acc1 = __builtin_amdgcn_wmma_f32_16x16x32_bf16(false, af, false, b1, (short)0, acc1, false, false);
    acc2 = __builtin_amdgcn_wmma_f32_16x16x32_bf16(false, af, false, b2, (short)0, acc2, false, false);
    acc3 = __builtin_amdgcn_wmma_f32_16x16x32_bf16(false, af, false, b3, (short)0, acc3, false, false);
  }

  // C/D layout: VGPR r = row m0 + (lane<16 ? r : r+8), col = n + (lane&15)
  const int r0 = (lane >> 4) << 3;
  v8f accs[4] = {acc0, acc1, acc2, acc3};
#pragma unroll
  for (int j = 0; j < 4; ++j) {
    const int n = n0 + j * 16 + l15;
#pragma unroll
    for (int r = 0; r < 8; ++r) {
      const size_t idx = (size_t)(m0 + r0 + r) * N + n;
      float v = accs[j][r];
      if (HAS_ADD) v += add[idx];
      C[idx] = v;
    }
  }
}

// ---------------- depthwise causal conv(4) + SiLU ----------------
__global__ void conv_silu_kernel(const float* __restrict__ xz, const float* __restrict__ cw,
                                 const float* __restrict__ cb, float* __restrict__ uf,
                                 __bf16* __restrict__ ub) {
  const int idx = blockIdx.x * blockDim.x + threadIdx.x;
  if (idx >= MTOK * DINNER) return;
  const int d = idx & (DINNER - 1);
  const int m = idx / DINNER;
  const int t = m & (LL - 1);
  const int bi = m / LL;
  float acc = cb[d];
#pragma unroll
  for (int k = 0; k < DCONV; ++k) {
    const int tt = t + k - (DCONV - 1);
    if (tt >= 0)
      acc += xz[(size_t)(bi * LL + tt) * (2 * DINNER) + d] * cw[d * DCONV + k];
  }
  const float s = acc / (1.f + __expf(-acc));   // silu
  uf[idx] = s;
  ub[idx] = f2bf(s);
}

// ---------------- extract dt_r (bf16) from dbl (stride DBLP) ----------------
__global__ void dtr_kernel(const float* __restrict__ dbl, __bf16* __restrict__ dtr) {
  const int i = blockIdx.x * blockDim.x + threadIdx.x;
  if (i >= MTOK * DTRANK) return;
  const int m = i / DTRANK;
  const int r = i - m * DTRANK;
  dtr[i] = f2bf(dbl[(size_t)m * DBLP + r]);
}

// ---------------- softplus(dt_lin + b_dt), in place ----------------
__global__ void softplus_kernel(float* __restrict__ dt, const float* __restrict__ bdt) {
  const int i = blockIdx.x * blockDim.x + threadIdx.x;
  if (i >= MTOK * DINNER) return;
  const int d = i & (DINNER - 1);
  const float xv = dt[i] + bdt[d];
  dt[i] = (xv > 20.f) ? xv : log1pf(__expf(xv));
}

// ---------------- selective-state scan (sequential in L) ----------------
// One thread per (batch, channel); 16 states held in registers.
__global__ void scan_kernel(const float* __restrict__ dt, const float* __restrict__ uf,
                            const float* __restrict__ dbl, const float* __restrict__ A_log,
                            float* __restrict__ ys) {
  const int c = blockIdx.x * blockDim.x + threadIdx.x;   // 0..BB*DINNER-1
  if (c >= BB * DINNER) return;
  const int d  = c & (DINNER - 1);
  const int bi = c / DINNER;
  float A[DSTATE], h[DSTATE];
#pragma unroll
  for (int n = 0; n < DSTATE; ++n) {
    A[n] = -__expf(A_log[d * DSTATE + n]);
    h[n] = 0.f;
  }
  for (int t = 0; t < LL; ++t) {
    const size_t m   = (size_t)bi * LL + t;
    const float dtv  = dt[m * DINNER + d];
    const float uv   = uf[m * DINNER + d];
    const float* row = dbl + m * DBLP;
    const float dtu  = dtv * uv;
    float y = 0.f;
#pragma unroll
    for (int n = 0; n < DSTATE; ++n) {
      const float Bn = row[DTRANK + n];
      const float Cn = row[DTRANK + DSTATE + n];
      h[n] = __expf(dtv * A[n]) * h[n] + dtu * Bn;
      y += h[n] * Cn;
    }
    ys[m * DINNER + d] = y;
  }
}

// ---------------- y = (ys + u*D) * silu(z) -> bf16 ----------------
__global__ void gate_kernel(const float* __restrict__ ys, const float* __restrict__ uf,
                            const float* __restrict__ Dv, const float* __restrict__ xz,
                            __bf16* __restrict__ yb) {
  const int i = blockIdx.x * blockDim.x + threadIdx.x;
  if (i >= MTOK * DINNER) return;
  const int d = i & (DINNER - 1);
  const int m = i / DINNER;
  const float z  = xz[(size_t)m * (2 * DINNER) + DINNER + d];
  const float sz = z / (1.f + __expf(-z));
  yb[i] = f2bf((ys[i] + uf[i] * Dv[d]) * sz);
}

// ---------------- workspace layout (256B-aligned, bytes) ----------------
#define ALIGN256(x) (((x) + 255) & ~((size_t)255))
static constexpr size_t OFF_XN    = 0;
static constexpr size_t SZ_XN     = ALIGN256((size_t)MTOK * DMODEL * 2);
static constexpr size_t OFF_WIN   = OFF_XN + SZ_XN;
static constexpr size_t SZ_WIN    = ALIGN256((size_t)2 * DINNER * DMODEL * 2);
static constexpr size_t OFF_WXP   = OFF_WIN + SZ_WIN;
static constexpr size_t SZ_WXP    = ALIGN256((size_t)DBLP * DINNER * 2);   // padded
static constexpr size_t OFF_WDT   = OFF_WXP + SZ_WXP;
static constexpr size_t SZ_WDT    = ALIGN256((size_t)DINNER * DTRANK * 2);
static constexpr size_t OFF_WOUT  = OFF_WDT + SZ_WDT;
static constexpr size_t SZ_WOUT   = ALIGN256((size_t)DMODEL * DINNER * 2);
static constexpr size_t OFF_XZ    = OFF_WOUT + SZ_WOUT;
static constexpr size_t SZ_XZ     = ALIGN256((size_t)MTOK * 2 * DINNER * 4);
static constexpr size_t OFF_UF    = OFF_XZ + SZ_XZ;
static constexpr size_t SZ_UF     = ALIGN256((size_t)MTOK * DINNER * 4);
static constexpr size_t OFF_UB    = OFF_UF + SZ_UF;
static constexpr size_t SZ_UB     = ALIGN256((size_t)MTOK * DINNER * 2);
static constexpr size_t OFF_DBL   = OFF_UB + SZ_UB;
static constexpr size_t SZ_DBL    = ALIGN256((size_t)MTOK * DBLP * 4);     // padded
static constexpr size_t OFF_DTR   = OFF_DBL + SZ_DBL;
static constexpr size_t SZ_DTR    = ALIGN256((size_t)MTOK * DTRANK * 2);
static constexpr size_t OFF_DT    = OFF_DTR + SZ_DTR;
static constexpr size_t SZ_DT     = ALIGN256((size_t)MTOK * DINNER * 4);
static constexpr size_t OFF_YS    = OFF_DT + SZ_DT;
static constexpr size_t SZ_YS     = ALIGN256((size_t)MTOK * DINNER * 4);
static constexpr size_t OFF_YB    = OFF_YS + SZ_YS;

extern "C" void kernel_launch(void* const* d_in, const int* in_sizes, int n_in,
                              void* d_out, int out_size, void* d_ws, size_t ws_size,
                              hipStream_t stream) {
  const float* x      = (const float*)d_in[0];
  const float* ln_w   = (const float*)d_in[1];
  const float* ln_b   = (const float*)d_in[2];
  const float* W_in   = (const float*)d_in[3];
  const float* conv_w = (const float*)d_in[4];
  const float* conv_b = (const float*)d_in[5];
  const float* W_xprj = (const float*)d_in[6];
  const float* W_dt   = (const float*)d_in[7];
  const float* b_dt   = (const float*)d_in[8];
  const float* A_log  = (const float*)d_in[9];
  const float* Dv     = (const float*)d_in[10];
  const float* W_out  = (const float*)d_in[11];
  float* out = (float*)d_out;

  char* ws = (char*)d_ws;
  __bf16* xn_b   = (__bf16*)(ws + OFF_XN);
  __bf16* win_b  = (__bf16*)(ws + OFF_WIN);
  __bf16* wxp_b  = (__bf16*)(ws + OFF_WXP);
  __bf16* wdt_b  = (__bf16*)(ws + OFF_WDT);
  __bf16* wout_b = (__bf16*)(ws + OFF_WOUT);
  float*  xz     = (float*)(ws + OFF_XZ);
  float*  uf     = (float*)(ws + OFF_UF);
  __bf16* ub     = (__bf16*)(ws + OFF_UB);
  float*  dbl    = (float*)(ws + OFF_DBL);
  __bf16* dtr    = (__bf16*)(ws + OFF_DTR);
  float*  dt     = (float*)(ws + OFF_DT);
  float*  ys     = (float*)(ws + OFF_YS);
  __bf16* yb     = (__bf16*)(ws + OFF_YB);

  const int TPB = 256;
  // 1) layernorm -> bf16
  ln_kernel<<<MTOK, TPB, 0, stream>>>(x, ln_w, ln_b, xn_b);
  // 2) weight conversions (x-proj zero-padded from 96 to 128 rows)
  {
    const int n = 2 * DINNER * DMODEL;
    cvt_bf16_kernel<<<(n + TPB - 1) / TPB, TPB, 0, stream>>>(W_in, win_b, n, n);
  }
  {
    const int ns = DBLW * DINNER, np = DBLP * DINNER;
    cvt_bf16_kernel<<<(np + TPB - 1) / TPB, TPB, 0, stream>>>(W_xprj, wxp_b, ns, np);
  }
  {
    const int n = DINNER * DTRANK;
    cvt_bf16_kernel<<<(n + TPB - 1) / TPB, TPB, 0, stream>>>(W_dt, wdt_b, n, n);
  }
  {
    const int n = DMODEL * DINNER;
    cvt_bf16_kernel<<<(n + TPB - 1) / TPB, TPB, 0, stream>>>(W_out, wout_b, n, n);
  }

  // 3) xz = xn @ W_in^T  (M=4096, N=4096, K=1024) -> 16384 wave-tiles
  gemm_wmma_bf16<false><<<(MTOK / 16) * ((2 * DINNER) / 64) / 8, TPB, 0, stream>>>(
      xn_b, win_b, xz, nullptr, MTOK, 2 * DINNER, DMODEL);

  // 4) depthwise conv + silu
  conv_silu_kernel<<<(MTOK * DINNER + TPB - 1) / TPB, TPB, 0, stream>>>(xz, conv_w, conv_b, uf, ub);

  // 5) dbl = u @ W_xproj_pad^T  (M=4096, N=128, K=2048)
  gemm_wmma_bf16<false><<<(MTOK / 16) * (DBLP / 64) / 8, TPB, 0, stream>>>(
      ub, wxp_b, dbl, nullptr, MTOK, DBLP, DINNER);

  // 6) dt_r -> bf16
  dtr_kernel<<<(MTOK * DTRANK + TPB - 1) / TPB, TPB, 0, stream>>>(dbl, dtr);

  // 7) dt_lin = dt_r @ W_dt^T  (M=4096, N=2048, K=64)
  gemm_wmma_bf16<false><<<(MTOK / 16) * (DINNER / 64) / 8, TPB, 0, stream>>>(
      dtr, wdt_b, dt, nullptr, MTOK, DINNER, DTRANK);

  // 8) dt = softplus(dt_lin + b_dt)
  softplus_kernel<<<(MTOK * DINNER + TPB - 1) / TPB, TPB, 0, stream>>>(dt, b_dt);

  // 9) selective scan (sequential over L)
  scan_kernel<<<(BB * DINNER + TPB - 1) / TPB, TPB, 0, stream>>>(dt, uf, dbl, A_log, ys);

  // 10) gate: y = (ys + u*D) * silu(z) -> bf16
  gate_kernel<<<(MTOK * DINNER + TPB - 1) / TPB, TPB, 0, stream>>>(ys, uf, Dv, xz, yb);

  // 11) out = x + y @ W_out^T  (M=4096, N=1024, K=2048), residual fused
  gemm_wmma_bf16<true><<<(MTOK / 16) * (DMODEL / 64) / 8, TPB, 0, stream>>>(
      yb, wout_b, out, x, MTOK, DMODEL, DINNER);

  (void)in_sizes; (void)n_in; (void)out_size; (void)ws_size;
}
